// DPAP2PNet_15874199126547
// MI455X (gfx1250) — compile-verified, matrix operand order checked by reading, not run on hardware
//
#include <hip/hip_runtime.h>

// ---------------------------------------------------------------------------
// DPAP2PNet forward for MI455X (gfx1250, wave32).
// All GEMM-like work (MLPs, 3x3 convs as 9-tap implicit GEMM) runs on
// v_wmma_f32_16x16x32_bf16.  Weight (B) tiles are staged in LDS once per
// workgroup (8 waves share them) to lift arithmetic intensity vs L2.
// Gathers (grid_sample / resize) are fp32 VALU.
// ---------------------------------------------------------------------------

typedef __attribute__((ext_vector_type(16))) __bf16 v16bf;
typedef __attribute__((ext_vector_type(8)))  float  v8f;

#define DEV __device__ __forceinline__

static constexpr int BB   = 8;
static constexpr int HID  = 256;
static constexpr int NPTS = 8192;          // B * 32 * 32

DEV unsigned short f2bf(float f) {           // round-to-nearest-even f32->bf16
  unsigned int u = __float_as_uint(f);
  u += 0x7fffu + ((u >> 16) & 1u);
  return (unsigned short)(u >> 16);
}
DEV float bf2f(unsigned short h) { return __uint_as_float(((unsigned int)h) << 16); }

// Load one 16x32 bf16 WMMA fragment (A row-major row, or B = weight row [out,K]).
// p points at element (row, k0 + kb) where kb = (lane>>4)*8.  Layout per ISA:
// lane holds K = kb..kb+7 then K = 16+kb..16+kb+7.  Works for global or LDS.
DEV v16bf load_frag(const unsigned short* p) {
  union { uint4 q[2]; v16bf v; } u;
  u.q[0] = *reinterpret_cast<const uint4*>(p);
  u.q[1] = *reinterpret_cast<const uint4*>(p + 16);
  return u.v;
}

// Bilinear sample with zeros padding, align_corners=True (x,y already scaled).
DEV float gs_plane(const float* base, int Hf, int Wf, float x, float y) {
  float x0f = floorf(x), y0f = floorf(y);
  int x0 = (int)x0f, y0 = (int)y0f, x1 = x0 + 1, y1 = y0 + 1;
  float wx = x - x0f, wy = y - y0f;
  bool vx0 = (x0 >= 0) && (x0 < Wf), vx1 = (x1 >= 0) && (x1 < Wf);
  bool vy0 = (y0 >= 0) && (y0 < Hf), vy1 = (y1 >= 0) && (y1 < Hf);
  float v00 = (vy0 && vx0) ? base[y0 * Wf + x0] : 0.f;
  float v01 = (vy0 && vx1) ? base[y0 * Wf + x1] : 0.f;
  float v10 = (vy1 && vx0) ? base[y1 * Wf + x0] : 0.f;
  float v11 = (vy1 && vx1) ? base[y1 * Wf + x1] : 0.f;
  return v00 * (1.f - wy) * (1.f - wx) + v01 * (1.f - wy) * wx +
         v10 * wy * (1.f - wx) + v11 * wy * wx;
}

// ----------------------------- packing kernels -----------------------------

__global__ void cvt_bf16_kernel(const float* __restrict__ s,
                                unsigned short* __restrict__ d, int n) {
  int t = blockIdx.x * 256 + threadIdx.x;
  if (t < n) d[t] = f2bf(s[t]);
}

// OIHW (3x3) -> [tap][O][I] bf16 so B-fragment rows are contiguous in K.
__global__ void pack_oihw3x3_kernel(const float* __restrict__ src,
                                    unsigned short* __restrict__ dst,
                                    int O, int I, int n) {
  int t = blockIdx.x * 256 + threadIdx.x;
  if (t >= n) return;
  int i = t % I; int rest = t / I; int o = rest % O; int tap = rest / O;
  dst[((size_t)tap * O + o) * I + i] = f2bf(src[((size_t)o * I + i) * 9 + tap]);
}

// ----------------------------- sampling kernels ----------------------------

// roi0[p][c] = bilinear(f0, anchor_p)
__global__ void sample_anchors_kernel(const float* __restrict__ f0,
                                      unsigned short* __restrict__ roi0) {
  int tid = blockIdx.x * 256 + threadIdx.x;       // 8192 * 256 threads
  int p = tid & (NPTS - 1); int c = tid >> 13;
  int b = p >> 10, iy = (p >> 5) & 31, ix = p & 31;
  float px = ix * 16.f + 8.f, py = iy * 16.f + 8.f;
  const float* plane = f0 + ((size_t)(b * HID + c)) * 128 * 128;
  float v = gs_plane(plane, 128, 128, px * (127.f / 512.f), py * (127.f / 512.f));
  roi0[(size_t)p * HID + c] = f2bf(v);
}

// deform MLP second layer (256 -> 2) + add anchor -> dprops (fp32)
__global__ void deform_epilogue_kernel(const unsigned short* __restrict__ hid,
                                       const float* __restrict__ w2,
                                       const float* __restrict__ b2,
                                       float* __restrict__ dprops) {
  int p = blockIdx.x * 256 + threadIdx.x;
  if (p >= NPTS) return;
  const unsigned short* h = hid + (size_t)p * HID;
  float o0 = b2[0], o1 = b2[1];
  for (int k = 0; k < HID; ++k) {
    float hv = bf2f(h[k]);
    o0 = fmaf(hv, w2[k], o0);
    o1 = fmaf(hv, w2[HID + k], o1);
  }
  int ix = p & 31, iy = (p >> 5) & 31;
  dprops[p * 2 + 0] = ix * 16.f + 8.f + o0;
  dprops[p * 2 + 1] = iy * 16.f + 8.f + o1;
}

// Multi-level sample at deformed points into padded NHWC bf16 [B][34][34][768].
__global__ void sample_refine_kernel(const float* __restrict__ f0,
                                     const float* __restrict__ f1,
                                     const float* __restrict__ f2,
                                     const float* __restrict__ dprops,
                                     unsigned short* __restrict__ PR) {
  int tid = blockIdx.x * 256 + threadIdx.x;       // 8*34*34*768 threads
  int c = tid % 768; int rest = tid / 768;
  int x34 = rest % 34; rest /= 34; int y34 = rest % 34; int b = rest / 34;
  unsigned short out = 0;
  if (x34 >= 1 && x34 <= 32 && y34 >= 1 && y34 <= 32) {
    int p = b * 1024 + (y34 - 1) * 32 + (x34 - 1);
    float px = dprops[p * 2], py = dprops[p * 2 + 1];
    int lvl = c >> 8, cc = c & 255;
    const float* f; int S; float sc;
    if (lvl == 0)      { f = f0; S = 128; sc = 127.f / 512.f; }
    else if (lvl == 1) { f = f1; S = 64;  sc = 63.f  / 512.f; }
    else               { f = f2; S = 32;  sc = 31.f  / 512.f; }
    float v = gs_plane(f + ((size_t)(b * HID + cc)) * S * S, S, S, px * sc, py * sc);
    out = f2bf(v);
  }
  PR[tid] = out;
}

// fm NCHW fp32 -> padded NHWC bf16 [B][130][130][256]
__global__ void pack_fm_kernel(const float* __restrict__ fm,
                               unsigned short* __restrict__ FMP) {
  int tid = blockIdx.x * 256 + threadIdx.x;       // 8*130*130*256 threads
  int c = tid & 255; int rest = tid >> 8;
  int x = rest % 130; rest /= 130; int y = rest % 130; int b = rest / 130;
  unsigned short o = 0;
  if (x >= 1 && x <= 128 && y >= 1 && y <= 128)
    o = f2bf(fm[(((size_t)b * HID + c) * 128 + (y - 1)) * 128 + (x - 1)]);
  FMP[tid] = o;
}

// ------------------------------ WMMA GEMMs ---------------------------------

// Out[M,256] = act(A[M,256] @ W[256,256]^T + bias), bf16 in/out, fp32 accum.
// Wave tile 16x64, block = 8 waves -> 128x64.  W tile staged in LDS (32KB).
__global__ __launch_bounds__(256)
void gemm_bf16_kernel(const unsigned short* __restrict__ A,
                      const unsigned short* __restrict__ W,
                      const float* __restrict__ bias,
                      unsigned short* __restrict__ Out,
                      int relu) {
  __shared__ __align__(16) unsigned short Bs[64 * 256];
  const int lane = threadIdx.x & 31, wave = threadIdx.x >> 5;
  const int m0 = blockIdx.y * 128 + wave * 16;
  const int n0 = blockIdx.x * 64;
  const int r = lane & 15, kb = (lane >> 4) * 8;

  // cooperative stage of weight rows n0..n0+63 (contiguous 32KB)
  {
    const uint4* src = reinterpret_cast<const uint4*>(W + (size_t)n0 * 256);
    uint4* dst = reinterpret_cast<uint4*>(Bs);
#pragma unroll
    for (int i = 0; i < 8; ++i) dst[threadIdx.x + i * 256] = src[threadIdx.x + i * 256];
  }
  __syncthreads();

  v8f acc[4];
#pragma unroll
  for (int t = 0; t < 4; ++t)
#pragma unroll
    for (int j = 0; j < 8; ++j) acc[t][j] = 0.f;

  const unsigned short* ap = A + (size_t)(m0 + r) * 256 + kb;
  for (int k0 = 0; k0 < 256; k0 += 32) {
    v16bf af = load_frag(ap + k0);
#pragma unroll
    for (int t = 0; t < 4; ++t) {
      v16bf bf = load_frag(Bs + (t * 16 + r) * 256 + k0 + kb);
      acc[t] = __builtin_amdgcn_wmma_f32_16x16x32_bf16(
          false, af, false, bf, (short)0, acc[t], false, false);
    }
  }
  const int mrow = m0 + ((lane >> 4) << 3);
#pragma unroll
  for (int t = 0; t < 4; ++t) {
    int n = n0 + t * 16 + r;
    float bv = bias[n];
#pragma unroll
    for (int j = 0; j < 8; ++j) {
      float v = acc[t][j] + bv;
      if (relu) v = fmaxf(v, 0.f);
      Out[(size_t)(mrow + j) * HID + n] = f2bf(v);
    }
  }
}

// refine conv: 9-tap implicit GEMM over PR[B][34][34][768], weights [9][256][768].
// Per tap the 64x768 weight slab is staged through LDS in three 32KB chunks.
__global__ __launch_bounds__(256)
void conv3x3_refine_kernel(const unsigned short* __restrict__ PR,
                           const unsigned short* __restrict__ CW,
                           const float* __restrict__ bias,
                           unsigned short* __restrict__ Out) {
  __shared__ __align__(16) unsigned short Bs[64 * 256];
  const int lane = threadIdx.x & 31, wave = threadIdx.x >> 5;
  const int mtile = blockIdx.y * 8 + wave;        // 0..511, 16 pixels each
  const int n0 = blockIdx.x * 64;
  const int r = lane & 15, kb = (lane >> 4) * 8;
  const int xt = mtile & 1, y = (mtile >> 1) & 31, b = mtile >> 6;
  const int x = xt * 16 + r;
  v8f acc[4];
#pragma unroll
  for (int t = 0; t < 4; ++t)
#pragma unroll
    for (int j = 0; j < 8; ++j) acc[t][j] = 0.f;

  for (int t9 = 0; t9 < 9; ++t9) {
    int dy = t9 / 3, dx = t9 % 3;
    const unsigned short* ap =
        PR + ((size_t)((b * 34 + y + dy) * 34 + (x + dx))) * 768 + kb;
    const unsigned short* wsrc = CW + (size_t)t9 * HID * 768 + (size_t)n0 * 768;
    for (int kc = 0; kc < 768; kc += 256) {
      __syncthreads();                             // previous chunk consumed
      // stage 64 rows x 256 K (row stride 768 in global)
      for (int i = threadIdx.x; i < 64 * 32; i += 256) {
        int row = i >> 5, col8 = i & 31;
        reinterpret_cast<uint4*>(Bs)[i] = *reinterpret_cast<const uint4*>(
            wsrc + (size_t)row * 768 + kc + col8 * 8);
      }
      __syncthreads();
      for (int k0 = 0; k0 < 256; k0 += 32) {
        __builtin_prefetch(ap + kc + k0 + 128, 0, 1);
        v16bf af = load_frag(ap + kc + k0);
#pragma unroll
        for (int t = 0; t < 4; ++t) {
          v16bf bf = load_frag(Bs + (t * 16 + r) * 256 + k0 + kb);
          acc[t] = __builtin_amdgcn_wmma_f32_16x16x32_bf16(
              false, af, false, bf, (short)0, acc[t], false, false);
        }
      }
    }
  }
  const int mrow = mtile * 16 + ((lane >> 4) << 3);
#pragma unroll
  for (int t = 0; t < 4; ++t) {
    int n = n0 + t * 16 + r;
    float bv = bias[n];
#pragma unroll
    for (int j = 0; j < 8; ++j)
      Out[(size_t)(mrow + j) * HID + n] = f2bf(acc[t][j] + bv);
  }
}

// mask conv: 9-tap implicit GEMM + fused BN + ReLU + 1x1 conv (channel sum).
// Block owns 16 pixels x all 256 channels (wave w -> channels [32w,32w+32)).
// Per tap the 256x256 weights are staged through LDS in 256x64 chunks (32KB).
__global__ __launch_bounds__(256)
void mask_conv_kernel(const unsigned short* __restrict__ FMP,
                      const unsigned short* __restrict__ MW,
                      const float* __restrict__ m1_b,
                      const float* __restrict__ bn_g, const float* __restrict__ bn_b,
                      const float* __restrict__ bn_m, const float* __restrict__ bn_v,
                      const float* __restrict__ m2_w, const float* __restrict__ m2_b,
                      float* __restrict__ Mbuf) {
  __shared__ __align__(16) unsigned short Bs[256 * 64];
  __shared__ float red[16];
  const int lane = threadIdx.x & 31, wave = threadIdx.x >> 5;
  const int n0 = wave * 32;
  const int mtile = blockIdx.x;                   // 0..8191
  const int xt = mtile & 7, y = (mtile >> 3) & 127, b = mtile >> 10;
  const int r = lane & 15, kb = (lane >> 4) * 8;
  const int x = xt * 16 + r;
  v8f acc[2];
#pragma unroll
  for (int t = 0; t < 2; ++t)
#pragma unroll
    for (int j = 0; j < 8; ++j) acc[t][j] = 0.f;

  for (int t9 = 0; t9 < 9; ++t9) {
    int dy = t9 / 3, dx = t9 % 3;
    const unsigned short* ap =
        FMP + ((size_t)((b * 130 + y + dy) * 130 + (x + dx))) * HID + kb;
    const unsigned short* wsrc = MW + (size_t)t9 * HID * HID;
    for (int kc = 0; kc < 256; kc += 64) {
      __syncthreads();
      // stage 256 rows x 64 K (row stride 256 in global, 64 in LDS)
      for (int i = threadIdx.x; i < 256 * 8; i += 256) {
        int row = i >> 3, col8 = i & 7;
        reinterpret_cast<uint4*>(Bs)[i] = *reinterpret_cast<const uint4*>(
            wsrc + (size_t)row * 256 + kc + col8 * 8);
      }
      __syncthreads();
      for (int k0 = 0; k0 < 64; k0 += 32) {
        __builtin_prefetch(ap + kc + k0 + 128, 0, 1);
        v16bf af = load_frag(ap + kc + k0);
#pragma unroll
        for (int t = 0; t < 2; ++t) {
          v16bf bf = load_frag(Bs + (n0 + t * 16 + r) * 64 + k0 + kb);
          acc[t] = __builtin_amdgcn_wmma_f32_16x16x32_bf16(
              false, af, false, bf, (short)0, acc[t], false, false);
        }
      }
    }
  }
  if (threadIdx.x < 16) red[threadIdx.x] = 0.f;
  __syncthreads();
  // per-channel BN scale/shift and 1x1 weight (2 channels per lane)
  float s[2], sh[2], ww[2];
#pragma unroll
  for (int t = 0; t < 2; ++t) {
    int n = n0 + t * 16 + r;
    s[t]  = rsqrtf(bn_v[n] + 1e-5f) * bn_g[n];
    sh[t] = bn_b[n] + (m1_b[n] - bn_m[n]) * s[t];
    ww[t] = m2_w[n];
  }
  const int mbase = (lane >> 4) << 3;
#pragma unroll
  for (int j = 0; j < 8; ++j) {
    float c = 0.f;
#pragma unroll
    for (int t = 0; t < 2; ++t) {
      float v = fmaxf(acc[t][j] * s[t] + sh[t], 0.f);
      c += v * ww[t];
    }
    atomicAdd(&red[mbase + j], c);
  }
  __syncthreads();
  if (threadIdx.x < 16)
    Mbuf[(size_t)(b * 130 + y + 1) * 130 + (xt * 16 + threadIdx.x + 1)] =
        red[threadIdx.x] + m2_b[0];
}

// ------------------------------- epilogues ---------------------------------

__global__ void init_mbuf_kernel(float* __restrict__ Mbuf,
                                 const float* __restrict__ m2_b, int n) {
  int t = blockIdx.x * 256 + threadIdx.x;
  if (t < n) Mbuf[t] = m2_b[0];                   // ring of 1x1 conv w/ pad=1
}

__global__ void heads_epilogue_kernel(const unsigned short* __restrict__ hr,
                                      const unsigned short* __restrict__ hc,
                                      const float* __restrict__ rw2,
                                      const float* __restrict__ rb2,
                                      const float* __restrict__ cw2,
                                      const float* __restrict__ cb2,
                                      const float* __restrict__ dprops,
                                      float* __restrict__ out) {
  int p = blockIdx.x * 256 + threadIdx.x;
  if (p >= NPTS) return;
  const unsigned short* h = hr + (size_t)p * HID;
  float o0 = rb2[0], o1 = rb2[1];
  for (int k = 0; k < HID; ++k) {
    float hv = bf2f(h[k]);
    o0 = fmaf(hv, rw2[k], o0);
    o1 = fmaf(hv, rw2[HID + k], o1);
  }
  out[p * 2 + 0] = dprops[p * 2 + 0] + o0;
  out[p * 2 + 1] = dprops[p * 2 + 1] + o1;
  float lg[6];
#pragma unroll
  for (int j = 0; j < 6; ++j) lg[j] = cb2[j];
  const unsigned short* g = hc + (size_t)p * HID;
  for (int k = 0; k < HID; ++k) {
    float hv = bf2f(g[k]);
#pragma unroll
    for (int j = 0; j < 6; ++j) lg[j] = fmaf(hv, cw2[j * HID + k], lg[j]);
  }
#pragma unroll
  for (int j = 0; j < 6; ++j) out[16384 + p * 6 + j] = lg[j];
}

// align_corners bilinear resize [8][130][130] -> out[8][512][512]
__global__ void resize_kernel(const float* __restrict__ Mbuf,
                              float* __restrict__ out) {
  int tid = blockIdx.x * 256 + threadIdx.x;       // 8*512*512 threads
  int b = tid >> 18; int oy = (tid >> 9) & 511; int ox = tid & 511;
  const float* mb = Mbuf + (size_t)b * 16900;
  float fy = oy * (129.f / 511.f);
  float y0f = floorf(fy); int y0 = (int)y0f; int y1 = min(y0 + 1, 129);
  float wy = fy - y0f;
  float fx = ox * (129.f / 511.f);
  float x0f = floorf(fx); int x0 = (int)x0f; int x1 = min(x0 + 1, 129);
  float wx = fx - x0f;
  float v = mb[y0 * 130 + x0] * (1.f - wy) * (1.f - wx) +
            mb[y0 * 130 + x1] * (1.f - wy) * wx +
            mb[y1 * 130 + x0] * wy * (1.f - wx) +
            mb[y1 * 130 + x1] * wy * wx;
  out[65536 + tid] = v;
}

// ------------------------------- launcher ----------------------------------

static inline unsigned char* carve(unsigned char*& p, size_t bytes) {
  unsigned char* r = p;
  p += (bytes + 255) & ~(size_t)255;
  return r;
}

extern "C" void kernel_launch(void* const* d_in, const int* in_sizes, int n_in,
                              void* d_out, int out_size, void* d_ws, size_t ws_size,
                              hipStream_t stream) {
  (void)in_sizes; (void)n_in; (void)out_size; (void)ws_size;
  const float* f0       = (const float*)d_in[1];
  const float* f1       = (const float*)d_in[2];
  const float* f2       = (const float*)d_in[3];
  const float* fm       = (const float*)d_in[4];
  const float* dw1      = (const float*)d_in[5];
  const float* db1      = (const float*)d_in[6];
  const float* dw2      = (const float*)d_in[7];
  const float* db2      = (const float*)d_in[8];
  const float* rw1      = (const float*)d_in[9];
  const float* rb1      = (const float*)d_in[10];
  const float* rw2      = (const float*)d_in[11];
  const float* rb2      = (const float*)d_in[12];
  const float* cw1      = (const float*)d_in[13];
  const float* cb1      = (const float*)d_in[14];
  const float* cw2      = (const float*)d_in[15];
  const float* cb2      = (const float*)d_in[16];
  const float* conv_w   = (const float*)d_in[17];
  const float* conv_b   = (const float*)d_in[18];
  const float* m1_w     = (const float*)d_in[19];
  const float* m1_b     = (const float*)d_in[20];
  const float* bn_g     = (const float*)d_in[21];
  const float* bn_b     = (const float*)d_in[22];
  const float* bn_m     = (const float*)d_in[23];
  const float* bn_v     = (const float*)d_in[24];
  const float* m2_w     = (const float*)d_in[25];
  const float* m2_b     = (const float*)d_in[26];
  float* out = (float*)d_out;

  unsigned char* p = (unsigned char*)d_ws;
  unsigned short* roi0  = (unsigned short*)carve(p, (size_t)NPTS * HID * 2);
  unsigned short* hidA  = (unsigned short*)carve(p, (size_t)NPTS * HID * 2);
  unsigned short* hidB  = (unsigned short*)carve(p, (size_t)NPTS * HID * 2);
  unsigned short* roir  = (unsigned short*)carve(p, (size_t)NPTS * HID * 2);
  unsigned short* w1d   = (unsigned short*)carve(p, (size_t)HID * HID * 2);
  unsigned short* w1r   = (unsigned short*)carve(p, (size_t)HID * HID * 2);
  unsigned short* w1c   = (unsigned short*)carve(p, (size_t)HID * HID * 2);
  unsigned short* cwP   = (unsigned short*)carve(p, (size_t)9 * HID * 768 * 2);
  unsigned short* m1P   = (unsigned short*)carve(p, (size_t)9 * HID * HID * 2);
  float*          dpr   = (float*)carve(p, (size_t)NPTS * 2 * 4);
  unsigned short* PR    = (unsigned short*)carve(p, (size_t)BB * 34 * 34 * 768 * 2);
  unsigned short* FMP   = (unsigned short*)carve(p, (size_t)BB * 130 * 130 * HID * 2);
  float*          Mbuf  = (float*)carve(p, (size_t)BB * 130 * 130 * 4);

  // --- weight packing ---
  cvt_bf16_kernel<<<256, 256, 0, stream>>>(dw1, w1d, HID * HID);
  cvt_bf16_kernel<<<256, 256, 0, stream>>>(rw1, w1r, HID * HID);
  cvt_bf16_kernel<<<256, 256, 0, stream>>>(cw1, w1c, HID * HID);
  pack_oihw3x3_kernel<<<6912, 256, 0, stream>>>(conv_w, cwP, HID, 768, 9 * HID * 768);
  pack_oihw3x3_kernel<<<2304, 256, 0, stream>>>(m1_w, m1P, HID, HID, 9 * HID * HID);

  // --- deform stage ---
  sample_anchors_kernel<<<8192, 256, 0, stream>>>(f0, roi0);
  gemm_bf16_kernel<<<dim3(4, 64), 256, 0, stream>>>(roi0, w1d, db1, hidA, 1);
  deform_epilogue_kernel<<<32, 256, 0, stream>>>(hidA, dw2, db2, dpr);

  // --- refine stage ---
  sample_refine_kernel<<<27744, 256, 0, stream>>>(f0, f1, f2, dpr, PR);
  conv3x3_refine_kernel<<<dim3(4, 64), 256, 0, stream>>>(PR, cwP, conv_b, roir);
  gemm_bf16_kernel<<<dim3(4, 64), 256, 0, stream>>>(roir, w1r, rb1, hidA, 1);
  gemm_bf16_kernel<<<dim3(4, 64), 256, 0, stream>>>(roir, w1c, cb1, hidB, 1);
  heads_epilogue_kernel<<<32, 256, 0, stream>>>(hidA, hidB, rw2, rb2, cw2, cb2, dpr, out);

  // --- mask head ---
  pack_fm_kernel<<<135200, 256, 0, stream>>>(fm, FMP);
  init_mbuf_kernel<<<529, 256, 0, stream>>>(Mbuf, m2_b, BB * 130 * 130);
  mask_conv_kernel<<<8192, 256, 0, stream>>>(FMP, m1P, m1_b, bn_g, bn_b, bn_m, bn_v,
                                             m2_w, m2_b, Mbuf);
  resize_kernel<<<8192, 256, 0, stream>>>(Mbuf, out);
}
